// SecondaryStructurePredictor_65292092834434
// MI455X (gfx1250) — matrix operand order repaired and checked by reading, not compiled, not run on hardware
//
#include <hip/hip_runtime.h>
#include <hip/hip_bf16.h>
#include <math.h>

// ---------------------------------------------------------------------------
// Types
// ---------------------------------------------------------------------------
typedef __bf16 bf16t;
typedef __attribute__((ext_vector_type(16))) __bf16 v16bf;
typedef __attribute__((ext_vector_type(8)))  __bf16 v8bf;
typedef __attribute__((ext_vector_type(8)))  float  v8f;

union F8 { v8f v; float f[8]; };

constexpr int B_ = 2, S_ = 128, D_ = 256, H_ = 8, HD_ = 32;
constexpr int DFF_ = 1024, NL_ = 2, CB_ = 64;
constexpr int MP_ = B_ * S_ * S_;   // 32768 pair rows
constexpr int MS_ = B_ * S_;        // 256 seq rows

__device__ inline v8f zero8() {
  v8f z = {0.f, 0.f, 0.f, 0.f, 0.f, 0.f, 0.f, 0.f};
  return z;
}

// ---------------------------------------------------------------------------
// CDNA5 async global->LDS copy (ASYNCcnt-tracked), per ISA 08_async_tensor §4.
// VDST = per-lane LDS byte offset, VADDR = 64-bit global address.
// ---------------------------------------------------------------------------
__device__ inline unsigned lds_off_of(const void* p) {
  // generic pointer to LDS: low 32 bits are the LDS byte offset
  return (unsigned)(size_t)p;
}

__device__ inline void async_load16(unsigned ldsoff, const void* g) {
  unsigned long long ga = (unsigned long long)(size_t)g;
  asm volatile("global_load_async_to_lds_b128 %0, %1, off"
               :
               : "v"(ldsoff), "v"(ga)
               : "memory");
}

template <int N>
__device__ inline void wait_async() {
  asm volatile("s_wait_asynccnt %0" : : "i"(N) : "memory");
}

// ---------------------------------------------------------------------------
// WMMA fragment loaders (wave32; A and B share a contiguous k order, C/D layout per CDNA5 ISA 7.12.2)
// A: 16x32 bf16 (MxK). lane holds row m = lane%16;
//   elems i -> k = k0 + 16*(lane/16) + i
// B: 32x16 bf16 (KxN) given W as (N,K) row-major (y = x @ W^T):
//   lane holds col n = n0 + lane%16; elems h -> k = k0 + 16*(lane/16) + h
// C/D f32 16x16: vgpr r, lane l -> (M = r + 8*(l/16), N = l%16)
// ---------------------------------------------------------------------------
__device__ inline v16bf frag_a(const bf16t* A, int lda, int m0, int k0, int lane) {
  const bf16t* p = A + (size_t)(m0 + (lane & 15)) * lda + (k0 + ((lane >> 4) << 4));
  return *(const v16bf*)p;
}

__device__ inline v16bf frag_b(const bf16t* W, int ldw, int n0, int k0, int lane) {
  const bf16t* p = W + (size_t)(n0 + (lane & 15)) * ldw + (k0 + ((lane >> 4) << 4));
  return *(const v16bf*)p;   // 32B contiguous
}

__device__ inline v8f wmma_bf16(v16bf a, v16bf b, v8f c) {
  return __builtin_amdgcn_wmma_f32_16x16x32_bf16(false, a, false, b, (short)0, c,
                                                 false, false);
}

// ---------------------------------------------------------------------------
// Generic GEMM: out = act(A @ W^T + bias), A (M,K) bf16, W (N,K) bf16.
// Block: 256 thr = 8 waves; block tile 64(M) x 128(N); wave tile 16x64.
// A/B tiles double-buffered in LDS via async global->LDS copies.
// Requires M % 64 == 0, K % 32 == 0, N % 16 == 0 (waves with n0 >= N idle).
// ---------------------------------------------------------------------------
enum { EPI_F32 = 0, EPI_BF16 = 1, EPI_PAIR = 2, EPI_RES = 3 };
enum { ACT_NONE = 0, ACT_RELU = 1, ACT_GELU = 2 };

template <int EPI, int ACT>
__global__ __launch_bounds__(256) void gemm_k(
    const bf16t* __restrict__ A, const bf16t* __restrict__ W,
    const float* __restrict__ bias, void* __restrict__ outp,
    float* __restrict__ aux, int M, int N, int K, float auxScale) {
  constexpr int PITCH = 48;  // halves; 96B rows: 32B-aligned, bank-spread
  __shared__ __align__(32) bf16t sA[2][64 * PITCH];
  __shared__ __align__(32) bf16t sB[2][128 * PITCH];

  const int tid = threadIdx.x;
  const int wv = tid >> 5, lane = tid & 31;
  const int m0b = blockIdx.y * 64;
  const int nb0 = blockIdx.x * 128;

  // staging assignment: each lane copies one 16B chunk of A, two of B
  const int arow = tid >> 2;        // 0..63
  const int acol = (tid & 3) * 8;   // halves (16B chunk)

  auto stage = [&](int buf, int k0) {
    async_load16(lds_off_of(&sA[buf][arow * PITCH + acol]),
                 A + (size_t)(m0b + arow) * K + k0 + acol);
#pragma unroll
    for (int c = 0; c < 2; ++c) {
      const int br = (tid >> 2) + 64 * c;   // 0..127
      int rr = nb0 + br;
      if (rr >= N) rr = 0;                  // clamp OOB weight rows
      async_load16(lds_off_of(&sB[buf][br * PITCH + acol]),
                   W + (size_t)rr * K + k0 + acol);
    }
  };

  F8 acc[4];
#pragma unroll
  for (int t = 0; t < 4; ++t) acc[t].v = zero8();

  const int KS = K >> 5;
  stage(0, 0);
  for (int ks = 0; ks < KS; ++ks) {
    const int cur = ks & 1;
    if (ks + 1 < KS) {
      stage(cur ^ 1, (ks + 1) << 5);  // prefetch next tile (3 asyncs in flight)
      wait_async<3>();                // current tile's copies are done
    } else {
      wait_async<0>();
    }
    __syncthreads();

    const bf16t* pA = &sA[cur][0];
    const bf16t* pB = &sB[cur][0];
    v16bf a = frag_a(pA, PITCH, (wv >> 1) * 16, 0, lane);
#pragma unroll
    for (int t = 0; t < 4; ++t) {
      const int nl = (wv & 1) * 64 + t * 16;
      if (nb0 + nl < N) {
        v16bf b = frag_b(pB, PITCH, nl, 0, lane);
        acc[t].v = wmma_bf16(a, b, acc[t].v);
      }
    }
    __syncthreads();  // protect buffer about to be overwritten
  }

  const int m0 = m0b + (wv >> 1) * 16;
  const int nb = nb0 + (wv & 1) * 64;
#pragma unroll
  for (int t = 0; t < 4; ++t) {
    const int n0 = nb + t * 16;
    if (n0 >= N) continue;
    const int cn = n0 + (lane & 15);
    const float bv = bias ? bias[cn] : 0.f;
#pragma unroll
    for (int r = 0; r < 8; ++r) {
      const int cm = m0 + r + ((lane >> 4) << 3);
      float v = acc[t].f[r] + bv;
      if (ACT == ACT_RELU) v = fmaxf(v, 0.f);
      if (ACT == ACT_GELU) v = 0.5f * v * (1.f + erff(v * 0.70710678118654752f));
      const size_t oi = (size_t)cm * N + cn;
      if (EPI == EPI_F32) {
        ((float*)outp)[oi] = v;
      } else if (EPI == EPI_BF16) {
        ((bf16t*)outp)[oi] = (bf16t)v;
      } else if (EPI == EPI_RES) {
        ((float*)outp)[oi] += v;
      } else {  // EPI_PAIR: pair += po, seq += po/S (mean over j)
        ((float*)outp)[oi] += v;
        atomicAdd(&aux[(size_t)(cm >> 7) * N + cn], v * auxScale);
      }
    }
  }
}

// ---------------------------------------------------------------------------
// Fused scaled-dot-product attention for one (slab, head):
//   scores = Q K^T / sqrt(32); softmax over keys; ctx = P V.
// Q/K/V/O are (nslab*S, ld*) bf16 with per-head column offset (c + h*HD).
// One block per (slab, head); 8 waves each own 16 query rows.
// ---------------------------------------------------------------------------
__global__ __launch_bounds__(256) void attn_k(
    const bf16t* __restrict__ Q, const bf16t* __restrict__ K,
    const bf16t* __restrict__ V, bf16t* __restrict__ O,
    int ldq, int ldk, int ldv, int ldo, int qc, int kc, int vc, int oc) {
  constexpr int PIT = 144;  // 288B pitch: 32B aligned, spreads LDS banks
  __shared__ __align__(32) bf16t lds_vt[HD_ * PIT];       // V transposed (d, kj)
  __shared__ __align__(32) bf16t lds_p[8 * 16 * PIT];     // probs, 16 rows/wave

  const int slab = blockIdx.x, h = blockIdx.y;
  const bf16t* Qb = Q + (size_t)slab * S_ * ldq + qc + h * HD_;
  const bf16t* Kb = K + (size_t)slab * S_ * ldk + kc + h * HD_;
  const bf16t* Vb = V + (size_t)slab * S_ * ldv + vc + h * HD_;
  bf16t* Ob = O + (size_t)slab * S_ * ldo + oc + h * HD_;

  const int tid = threadIdx.x;
  for (int idx = tid; idx < S_ * HD_; idx += 256) {
    const int kj = idx >> 5, d = idx & (HD_ - 1);
    lds_vt[d * PIT + kj] = Vb[(size_t)kj * ldv + d];
  }
  __syncthreads();

  const int wv = tid >> 5, lane = tid & 31;
  const int m0 = wv * 16;

  // scores: Q(16x32) x K^T -> 8 tiles of 16x16, one WMMA each (HD == 32)
  v16bf a = frag_a(Qb, ldq, m0, 0, lane);
  F8 sc[8];
#pragma unroll
  for (int t = 0; t < 8; ++t) {
    v16bf b = frag_b(Kb, ldk, t * 16, 0, lane);
    sc[t].v = wmma_bf16(a, b, zero8());
  }

  // softmax over 128 keys. Row (r, lane/16) spans 16 lanes of a half-wave
  // (cols = lane%16) x 8 register tiles. Reduce across t then shfl_xor 8/4/2/1.
  const float scale = 0.17677669529663687f;  // 1/sqrt(32)
#pragma unroll
  for (int r = 0; r < 8; ++r) {
    float mx = -1e30f;
#pragma unroll
    for (int t = 0; t < 8; ++t) { sc[t].f[r] *= scale; mx = fmaxf(mx, sc[t].f[r]); }
#pragma unroll
    for (int msk = 8; msk >= 1; msk >>= 1) mx = fmaxf(mx, __shfl_xor(mx, msk, 32));
    float sum = 0.f;
#pragma unroll
    for (int t = 0; t < 8; ++t) {
      const float e = __expf(sc[t].f[r] - mx);
      sc[t].f[r] = e;
      sum += e;
    }
#pragma unroll
    for (int msk = 8; msk >= 1; msk >>= 1) sum += __shfl_xor(sum, msk, 32);
    const float inv = 1.f / sum;
#pragma unroll
    for (int t = 0; t < 8; ++t) sc[t].f[r] *= inv;
  }

  // relayout probs C-layout -> A-layout via LDS
#pragma unroll
  for (int t = 0; t < 8; ++t)
#pragma unroll
    for (int r = 0; r < 8; ++r) {
      const int row = r + ((lane >> 4) << 3);
      const int col = t * 16 + (lane & 15);
      lds_p[(m0 + row) * PIT + col] = (bf16t)sc[t].f[r];
    }
  __syncthreads();

  // ctx = P(16x128) @ V(128x32): 4 k-steps x 2 n-tiles
  F8 o2[2];
  o2[0].v = zero8();
  o2[1].v = zero8();
  const bf16t* Pw = lds_p + m0 * PIT;
#pragma unroll
  for (int k0 = 0; k0 < S_; k0 += 32) {
    v16bf a2 = frag_a(Pw, PIT, 0, k0, lane);
#pragma unroll
    for (int t = 0; t < 2; ++t) {
      v16bf b = frag_b(lds_vt, PIT, t * 16, k0, lane);
      o2[t].v = wmma_bf16(a2, b, o2[t].v);
    }
  }
#pragma unroll
  for (int t = 0; t < 2; ++t)
#pragma unroll
    for (int r = 0; r < 8; ++r) {
      const int j = m0 + r + ((lane >> 4) << 3);
      const int d = t * 16 + (lane & 15);
      Ob[(size_t)j * ldo + d] = (bf16t)o2[t].f[r];
    }
}

// ---------------------------------------------------------------------------
// LayerNorm over D=256; one block (256 thr) per row; fp32 and/or bf16 out.
// ---------------------------------------------------------------------------
__global__ __launch_bounds__(256) void ln_k(
    const float* __restrict__ x, const float* __restrict__ g,
    const float* __restrict__ bta, float* __restrict__ y32,
    bf16t* __restrict__ y16) {
  __shared__ float red[256];
  const int row = blockIdx.x, d = threadIdx.x;
  const float v = x[(size_t)row * D_ + d];
  red[d] = v;
  __syncthreads();
  for (int s = 128; s > 0; s >>= 1) { if (d < s) red[d] += red[d + s]; __syncthreads(); }
  const float mean = red[0] * (1.f / D_);
  __syncthreads();
  const float c = v - mean;
  red[d] = c * c;
  __syncthreads();
  for (int s = 128; s > 0; s >>= 1) { if (d < s) red[d] += red[d + s]; __syncthreads(); }
  const float var = red[0] * (1.f / D_);
  const float o = c * rsqrtf(var + 1e-5f) * g[d] + bta[d];
  if (y32) y32[(size_t)row * D_ + d] = o;
  if (y16) y16[(size_t)row * D_ + d] = (bf16t)o;
}

// ---------------------------------------------------------------------------
// Elementwise kernels
// ---------------------------------------------------------------------------
__global__ void conv_k(const float* __restrict__ s, bf16t* __restrict__ dst, int n) {
  for (int i = blockIdx.x * blockDim.x + threadIdx.x; i < n; i += gridDim.x * blockDim.x)
    dst[i] = (bf16t)s[i];
}

__global__ void pairinit_k(const float* __restrict__ pinit, float* __restrict__ pair, int n) {
  for (int i = blockIdx.x * blockDim.x + threadIdx.x; i < n; i += gridDim.x * blockDim.x)
    pair[i] = pinit[i & (D_ - 1)];
}

__global__ void copy_k(const float* __restrict__ s, float* __restrict__ dst, int n) {
  for (int i = blockIdx.x * blockDim.x + threadIdx.x; i < n; i += gridDim.x * blockDim.x)
    dst[i] = s[i];
}

// comb[b,i,j,:] = seqn[b,i,:] + seqn[b,j,:] + pair[b,i,j,:]  (bf16 out)
__global__ void comb_k(const float* __restrict__ seqn, const float* __restrict__ pair,
                       bf16t* __restrict__ comb, int total) {
  for (int idx = blockIdx.x * blockDim.x + threadIdx.x; idx < total;
       idx += gridDim.x * blockDim.x) {
    const int row = idx >> 8;                        // / D
    const int dd = idx & 255;
    const int bi = row >> 7;                         // b*S + i
    const int bj = ((row >> 14) << 7) | (row & 127); // b*S + j
    comb[idx] = (bf16t)(seqn[(size_t)bi * D_ + dd] + seqn[(size_t)bj * D_ + dd] +
                        pair[idx]);
  }
}

// Final tiny head: (MP,64) bf16 @ p3w^T(3,64) + p3b -> (MP,3) fp32
__global__ __launch_bounds__(256) void pk3_k(const bf16t* __restrict__ h2,
                                             const float* __restrict__ w,
                                             const float* __restrict__ b,
                                             float* __restrict__ out) {
  const int m = blockIdx.x * blockDim.x + threadIdx.x;
  if (m >= MP_) return;
  float s0 = b[0], s1 = b[1], s2 = b[2];
  const bf16t* hp = h2 + (size_t)m * CB_;
  for (int k = 0; k < CB_; ++k) {
    const float x = (float)hp[k];
    s0 += x * w[k];
    s1 += x * w[CB_ + k];
    s2 += x * w[2 * CB_ + k];
  }
  out[(size_t)m * 3 + 0] = s0;
  out[(size_t)m * 3 + 1] = s1;
  out[(size_t)m * 3 + 2] = s2;
}

// ---------------------------------------------------------------------------
// Host orchestration
// ---------------------------------------------------------------------------
extern "C" void kernel_launch(void* const* d_in, const int* in_sizes, int n_in,
                              void* d_out, int out_size, void* d_ws, size_t ws_size,
                              hipStream_t stream) {
  (void)in_sizes; (void)n_in; (void)out_size; (void)ws_size;

  const float* embeddings = (const float*)d_in[0];
  const float* inp_w  = (const float*)d_in[1];
  const float* inp_b  = (const float*)d_in[2];
  const float* pinit  = (const float*)d_in[3];
  const float* pa_qw  = (const float*)d_in[4];
  const float* pa_qb  = (const float*)d_in[5];
  const float* pa_kw  = (const float*)d_in[6];
  const float* pa_kb  = (const float*)d_in[7];
  const float* pa_vw  = (const float*)d_in[8];
  const float* pa_vb  = (const float*)d_in[9];
  const float* pa_ow  = (const float*)d_in[10];
  const float* pa_ob  = (const float*)d_in[11];
  const float* pn_g   = (const float*)d_in[12];
  const float* pn_b   = (const float*)d_in[13];
  const float* mha_in_w  = (const float*)d_in[14];
  const float* mha_in_b  = (const float*)d_in[15];
  const float* mha_out_w = (const float*)d_in[16];
  const float* mha_out_b = (const float*)d_in[17];
  const float* sn_g   = (const float*)d_in[18];
  const float* sn_b   = (const float*)d_in[19];
  const float* ff_w1  = (const float*)d_in[20];
  const float* ff_b1  = (const float*)d_in[21];
  const float* ff_w2  = (const float*)d_in[22];
  const float* ff_b2  = (const float*)d_in[23];
  const float* fn_g   = (const float*)d_in[24];
  const float* fn_b   = (const float*)d_in[25];
  const float* c1w    = (const float*)d_in[26];
  const float* c1b    = (const float*)d_in[27];
  const float* c2w    = (const float*)d_in[28];
  const float* c2b    = (const float*)d_in[29];
  const float* p1w    = (const float*)d_in[30];
  const float* p1b    = (const float*)d_in[31];
  const float* p2w    = (const float*)d_in[32];
  const float* p2b    = (const float*)d_in[33];
  const float* p3w    = (const float*)d_in[34];
  const float* p3b    = (const float*)d_in[35];

  char* ws = (char*)d_ws;
  size_t off = 0;
  auto alloc = [&](size_t bytes) -> char* {
    char* p = ws + off;
    off = (off + bytes + 255) & ~(size_t)255;
    return p;
  };

  float* pair   = (float*)alloc((size_t)MP_ * D_ * 4);
  bf16t* comb   = (bf16t*)alloc((size_t)MP_ * D_ * 2);  // also ctx / pair_bf
  bf16t* qb     = (bf16t*)alloc((size_t)MP_ * D_ * 2);  // also contact h1
  bf16t* kb     = (bf16t*)alloc((size_t)MP_ * D_ * 2);  // also pk h1
  bf16t* vb     = (bf16t*)alloc((size_t)MP_ * D_ * 2);  // also pk h2
  float* seq    = (float*)alloc((size_t)MS_ * D_ * 4);
  float* seqn_f = (float*)alloc((size_t)MS_ * D_ * 4);
  bf16t* seqn_b = (bf16t*)alloc((size_t)MS_ * D_ * 2);
  bf16t* qkvs   = (bf16t*)alloc((size_t)MS_ * 3 * D_ * 2);
  bf16t* ctxs   = (bf16t*)alloc((size_t)MS_ * D_ * 2);
  bf16t* ffh    = (bf16t*)alloc((size_t)MS_ * DFF_ * 2);
  bf16t* embb   = (bf16t*)alloc((size_t)MS_ * D_ * 2);
  bf16t* w_inp  = (bf16t*)alloc((size_t)D_ * D_ * 2);
  bf16t* w_q    = (bf16t*)alloc((size_t)NL_ * D_ * D_ * 2);
  bf16t* w_k    = (bf16t*)alloc((size_t)NL_ * D_ * D_ * 2);
  bf16t* w_v    = (bf16t*)alloc((size_t)NL_ * D_ * D_ * 2);
  bf16t* w_o    = (bf16t*)alloc((size_t)NL_ * D_ * D_ * 2);
  bf16t* w_mi   = (bf16t*)alloc((size_t)NL_ * 3 * D_ * D_ * 2);
  bf16t* w_mo   = (bf16t*)alloc((size_t)NL_ * D_ * D_ * 2);
  bf16t* w_f1   = (bf16t*)alloc((size_t)NL_ * DFF_ * D_ * 2);
  bf16t* w_f2   = (bf16t*)alloc((size_t)NL_ * D_ * DFF_ * 2);
  bf16t* w_c1   = (bf16t*)alloc((size_t)128 * 256 * 2);
  bf16t* w_c2   = (bf16t*)alloc((size_t)64 * 128 * 2);
  bf16t* w_p1   = (bf16t*)alloc((size_t)128 * 256 * 2);
  bf16t* w_p2   = (bf16t*)alloc((size_t)64 * 128 * 2);

  auto conv = [&](const float* s, bf16t* dst, int n) {
    int blocks = (n + 255) / 256;
    if (blocks > 4096) blocks = 4096;
    conv_k<<<blocks, 256, 0, stream>>>(s, dst, n);
  };

  // Weight/activation conversion to bf16
  conv(embeddings, embb, MS_ * D_);
  conv(inp_w, w_inp, D_ * D_);
  conv(pa_qw, w_q, NL_ * D_ * D_);
  conv(pa_kw, w_k, NL_ * D_ * D_);
  conv(pa_vw, w_v, NL_ * D_ * D_);
  conv(pa_ow, w_o, NL_ * D_ * D_);
  conv(mha_in_w, w_mi, NL_ * 3 * D_ * D_);
  conv(mha_out_w, w_mo, NL_ * D_ * D_);
  conv(ff_w1, w_f1, NL_ * DFF_ * D_);
  conv(ff_w2, w_f2, NL_ * D_ * DFF_);
  conv(c1w, w_c1, 128 * 256);
  conv(c2w, w_c2, 64 * 128);
  conv(p1w, w_p1, 128 * 256);
  conv(p2w, w_p2, 64 * 128);

  pairinit_k<<<4096, 256, 0, stream>>>(pinit, pair, MP_ * D_);

  // seq = embeddings @ inp_w^T + inp_b
  gemm_k<EPI_F32, ACT_NONE><<<dim3(2, MS_ / 64), 256, 0, stream>>>(
      embb, w_inp, inp_b, seq, nullptr, MS_, D_, D_, 0.f);

  const dim3 gP((D_ + 127) / 128, MP_ / 64);  // (2, 512)
  const dim3 gS(2, MS_ / 64);                 // (2, 4)

  for (int l = 0; l < NL_; ++l) {
    // --- pair attention ---
    ln_k<<<MS_, 256, 0, stream>>>(seq, pn_g + l * D_, pn_b + l * D_, seqn_f, nullptr);
    comb_k<<<4096, 256, 0, stream>>>(seqn_f, pair, comb, MP_ * D_);

    gemm_k<EPI_BF16, ACT_NONE><<<gP, 256, 0, stream>>>(
        comb, w_q + (size_t)l * D_ * D_, pa_qb + l * D_, qb, nullptr, MP_, D_, D_, 0.f);
    gemm_k<EPI_BF16, ACT_NONE><<<gP, 256, 0, stream>>>(
        comb, w_k + (size_t)l * D_ * D_, pa_kb + l * D_, kb, nullptr, MP_, D_, D_, 0.f);
    gemm_k<EPI_BF16, ACT_NONE><<<gP, 256, 0, stream>>>(
        comb, w_v + (size_t)l * D_ * D_, pa_vb + l * D_, vb, nullptr, MP_, D_, D_, 0.f);

    // per (b,i) row-slab, per head; ctx overwrites comb
    attn_k<<<dim3(MS_, H_), 256, 0, stream>>>(qb, kb, vb, comb, D_, D_, D_, D_,
                                              0, 0, 0, 0);

    // out-proj fused: pair += po ; seq += mean_j(po)
    gemm_k<EPI_PAIR, ACT_NONE><<<gP, 256, 0, stream>>>(
        comb, w_o + (size_t)l * D_ * D_, pa_ob + l * D_, pair, seq, MP_, D_, D_,
        1.0f / S_);

    // --- seq MHA ---
    ln_k<<<MS_, 256, 0, stream>>>(seq, sn_g + l * D_, sn_b + l * D_, nullptr, seqn_b);
    gemm_k<EPI_BF16, ACT_NONE><<<dim3(6, MS_ / 64), 256, 0, stream>>>(
        seqn_b, w_mi + (size_t)l * 3 * D_ * D_, mha_in_b + l * 3 * D_, qkvs, nullptr,
        MS_, 3 * D_, D_, 0.f);
    attn_k<<<dim3(B_, H_), 256, 0, stream>>>(qkvs, qkvs, qkvs, ctxs,
                                             3 * D_, 3 * D_, 3 * D_, D_,
                                             0, D_, 2 * D_, 0);
    gemm_k<EPI_RES, ACT_NONE><<<gS, 256, 0, stream>>>(
        ctxs, w_mo + (size_t)l * D_ * D_, mha_out_b + l * D_, seq, nullptr,
        MS_, D_, D_, 0.f);

    // --- FFN ---
    ln_k<<<MS_, 256, 0, stream>>>(seq, fn_g + l * D_, fn_b + l * D_, nullptr, seqn_b);
    gemm_k<EPI_BF16, ACT_GELU><<<dim3(8, MS_ / 64), 256, 0, stream>>>(
        seqn_b, w_f1 + (size_t)l * DFF_ * D_, ff_b1 + l * DFF_, ffh, nullptr,
        MS_, DFF_, D_, 0.f);
    gemm_k<EPI_RES, ACT_NONE><<<gS, 256, 0, stream>>>(
        ffh, w_f2 + (size_t)l * D_ * DFF_, ff_b2 + l * D_, seq, nullptr,
        MS_, D_, DFF_, 0.f);
  }

  // --- heads ---
  conv(pair, comb, MP_ * D_);  // pair -> bf16 (comb is free now)

  float* contact = (float*)d_out;
  float* pk = contact + (size_t)MP_ * CB_;
  float* pair_out = pk + (size_t)MP_ * 3;

  gemm_k<EPI_BF16, ACT_RELU><<<dim3(1, MP_ / 64), 256, 0, stream>>>(
      comb, w_c1, c1b, qb, nullptr, MP_, 128, 256, 0.f);
  gemm_k<EPI_F32, ACT_NONE><<<dim3(1, MP_ / 64), 256, 0, stream>>>(
      qb, w_c2, c2b, contact, nullptr, MP_, 64, 128, 0.f);

  gemm_k<EPI_BF16, ACT_RELU><<<dim3(1, MP_ / 64), 256, 0, stream>>>(
      comb, w_p1, p1b, kb, nullptr, MP_, 128, 256, 0.f);
  gemm_k<EPI_BF16, ACT_RELU><<<dim3(1, MP_ / 64), 256, 0, stream>>>(
      kb, w_p2, p2b, vb, nullptr, MP_, 64, 128, 0.f);
  pk3_k<<<MP_ / 256, 256, 0, stream>>>(vb, p3w, p3b, pk);

  copy_k<<<4096, 256, 0, stream>>>(pair, pair_out, MP_ * D_);
}
